// Memory_91122026152462
// MI455X (gfx1250) — matrix operand order repaired and checked by reading, not compiled, not run on hardware
//
#include <hip/hip_runtime.h>
#include <math.h>

// MI455X (gfx1250): wave32, WMMA 16x16x32 bf16 with f32 accumulate for the
// dominant 154.6 GFLOP extractor GEMM; everything else is HBM-bound VALU.

#define DV   768
#define K2   1536
#define CTOK 65536
#define NTOK 65536
#define MS   25

typedef __attribute__((ext_vector_type(16))) __bf16 v16bf;
typedef __attribute__((ext_vector_type(8)))  __bf16 v8bf;
typedef __attribute__((ext_vector_type(8)))  float  v8f;

__device__ __forceinline__ unsigned short f2bf(float f) {
  unsigned u = __float_as_uint(f);
  u += 0x7fffu + ((u >> 16) & 1u);           // round-to-nearest-even
  return (unsigned short)(u >> 16);
}
__device__ __forceinline__ float wredsum(float v) {
  v += __shfl_xor(v, 16, 32);
  v += __shfl_xor(v,  8, 32);
  v += __shfl_xor(v,  4, 32);
  v += __shfl_xor(v,  2, 32);
  v += __shfl_xor(v,  1, 32);
  return v;
}
// order-preserving float<->uint keys so atomicMax(uint) == float max
__device__ __forceinline__ unsigned fenc(float f) {
  unsigned u = __float_as_uint(f);
  return (u & 0x80000000u) ? ~u : (u | 0x80000000u);
}
__device__ __forceinline__ float fdec(unsigned k) {
  return __uint_as_float((k & 0x80000000u) ? (k & 0x7fffffffu) : ~k);
}

// ---------------- init: zero accumulators -------------------------------
__global__ __launch_bounds__(256) void kinit(float* seg, float* counts,
                                             float* loss, unsigned* colmax) {
  int t = threadIdx.x;
  for (int i = t; i < MS * DV; i += 256) seg[i] = 0.0f;
  if (t < MS) { counts[t] = 0.0f; colmax[t] = fenc(-__builtin_inff()); }
  if (t == 0) *loss = 0.0f;
}

// ---------------- W fp32 -> bf16 ----------------------------------------
__global__ __launch_bounds__(256) void kconvw(const float* __restrict__ w,
                                              unsigned short* __restrict__ wbf) {
  int i = blockIdx.x * 256 + threadIdx.x;
  if (i < DV * K2) wbf[i] = f2bf(w[i]);
}

// ---------------- read path: attention over cache, emit bf16 cat --------
// one wave per text token; 768 = 32 lanes x 24 elems
__global__ __launch_bounds__(256) void kread(const float* __restrict__ text,
                                             const float* __restrict__ cache,
                                             unsigned short* __restrict__ abf) {
  int lane = threadIdx.x & 31, wave = threadIdx.x >> 5;
  int c = blockIdx.x * 8 + wave;
  const float* row = text + (size_t)c * DV;
  float x[24]; float ss = 0.0f;
#pragma unroll
  for (int j = 0; j < 24; ++j) { x[j] = row[lane + 32 * j]; ss += x[j] * x[j]; }
  ss = wredsum(ss);
  float inv = 1.0f / fmaxf(sqrtf(ss), 1e-12f);

  float sc[MS]; float mx = -__builtin_inff();
#pragma unroll
  for (int s = 0; s < MS; ++s) {
    const float* cr = cache + (size_t)s * DV;
    float d = 0.0f;
#pragma unroll
    for (int j = 0; j < 24; ++j) d += x[j] * cr[lane + 32 * j];
    d = wredsum(d) * inv;
    sc[s] = d; mx = fmaxf(mx, d);
  }
  float se = 0.0f;
#pragma unroll
  for (int s = 0; s < MS; ++s) { sc[s] = expf(sc[s] - mx); se += sc[s]; }
  float rse = 1.0f / se;

  unsigned short* orow = abf + (size_t)c * K2;
#pragma unroll
  for (int j = 0; j < 24; ++j) {
    float f = 0.0f;
#pragma unroll
    for (int s = 0; s < MS; ++s) f += sc[s] * cache[(size_t)s * DV + lane + 32 * j];
    f *= rse;
    orow[lane + 32 * j]      = f2bf(x[j]);   // raw text goes into cat
    orow[DV + lane + 32 * j] = f2bf(f);      // fine feature
  }
}

// ---------------- big GEMM: out = 0.2*(cat @ W^T) + text  (WMMA bf16) ---
// wave -> 16 rows; block = 8 waves = 128 rows; 12 chunks of 64 cols; K=48x32
__global__ __launch_bounds__(256) void kgemm(const unsigned short* __restrict__ abf,
                                             const unsigned short* __restrict__ wbf,
                                             const float* __restrict__ text,
                                             float* __restrict__ out) {
  int lane = threadIdx.x & 31, wave = threadIdx.x >> 5;
  int m0 = blockIdx.x * 128 + wave * 16;
  int hs = (lane < 16) ? 0 : 1;              // K-half select per ISA layout
  int mrow = m0 + (lane & 15);
  const unsigned short* arow = abf + (size_t)mrow * K2 + (hs ? 8 : 0);

  for (int nc = 0; nc < 12; ++nc) {
    int n0 = nc * 64;
    v8f acc[4] = {};
    const unsigned short* bbase =
        wbf + (size_t)(n0 + (lane & 15)) * K2 + (hs ? 8 : 0);

    for (int kk = 0; kk < 48; ++kk) {
      int kb = kk * 32;
      v8bf alo = *reinterpret_cast<const v8bf*>(arow + kb);
      v8bf ahi = *reinterpret_cast<const v8bf*>(arow + kb + 16);
      v16bf av = __builtin_shufflevector(alo, ahi, 0,1,2,3,4,5,6,7,
                                                   8,9,10,11,12,13,14,15);
#pragma unroll
      for (int t = 0; t < 4; ++t) {
        const unsigned short* bp = bbase + (size_t)t * 16 * K2 + kb;
        v8bf bl = *reinterpret_cast<const v8bf*>(bp);
        v8bf bh = *reinterpret_cast<const v8bf*>(bp + 16);
        v16bf bv = __builtin_shufflevector(bl, bh, 0,1,2,3,4,5,6,7,
                                                   8,9,10,11,12,13,14,15);
        acc[t] = __builtin_amdgcn_wmma_f32_16x16x32_bf16(
            false, av, false, bv, (short)0, acc[t], false, false);
      }
    }
    // C/D layout: VGPR r -> M=r (lanes 0-15) or M=8+r (lanes 16-31), N=lane&15
#pragma unroll
    for (int t = 0; t < 4; ++t) {
      int gcol = n0 + t * 16 + (lane & 15);
#pragma unroll
      for (int r = 0; r < 8; ++r) {
        int grow = m0 + ((lane < 16) ? r : 8 + r);
        size_t o = (size_t)grow * DV + gcol;
        out[o] = 0.2f * acc[t][r] + text[o];
      }
    }
  }
}

// ---------------- loss: mean |normalize(out) - text| --------------------
__global__ __launch_bounds__(256) void kloss(const float* __restrict__ tff,
                                             const float* __restrict__ text,
                                             float* __restrict__ loss) {
  int lane = threadIdx.x & 31, wave = threadIdx.x >> 5;
  int r = blockIdx.x * 8 + wave;
  const float* row = tff + (size_t)r * DV;
  const float* trow = text + (size_t)r * DV;
  float y[24]; float ss = 0.0f;
#pragma unroll
  for (int j = 0; j < 24; ++j) { y[j] = row[lane + 32 * j]; ss += y[j] * y[j]; }
  ss = wredsum(ss);
  float inv = 1.0f / fmaxf(sqrtf(ss), 1e-12f);
  float s = 0.0f;
#pragma unroll
  for (int j = 0; j < 24; ++j) s += fabsf(y[j] * inv - trow[lane + 32 * j]);
  s = wredsum(s);
  if (lane == 0) atomicAdd(loss, s * (1.0f / ((float)CTOK * (float)DV)));
}

// ---------------- image pass 1: argmax slot + raw score + column max ----
__global__ __launch_bounds__(256) void kimg1(const float* __restrict__ image,
                                             const float* __restrict__ cache,
                                             int* __restrict__ idxArr,
                                             float* __restrict__ rowsc,
                                             unsigned* __restrict__ colmax) {
  __shared__ unsigned cmsh[MS];
  int lane = threadIdx.x & 31, wave = threadIdx.x >> 5;
  if (threadIdx.x < MS) cmsh[threadIdx.x] = fenc(-__builtin_inff());
  __syncthreads();

  int c = blockIdx.x * 8 + wave;
  const float* row = image + (size_t)c * DV;
  float x[24]; float ss = 0.0f;
#pragma unroll
  for (int j = 0; j < 24; ++j) { x[j] = row[lane + 32 * j]; ss += x[j] * x[j]; }
  ss = wredsum(ss);
  float inv = 1.0f / fmaxf(sqrtf(ss), 1e-12f);

  float best = -__builtin_inff(); int bj = 0;
  float sc[MS];
#pragma unroll
  for (int s = 0; s < MS; ++s) {
    const float* cr = cache + (size_t)s * DV;
    float d = 0.0f;
#pragma unroll
    for (int j = 0; j < 24; ++j) d += x[j] * cr[lane + 32 * j];
    d = wredsum(d) * inv;
    sc[s] = d;
    if (d > best) { best = d; bj = s; }      // first max wins (argmax semantics)
  }
  if (lane == 0) {
    idxArr[c] = bj;
    rowsc[c] = best;
#pragma unroll
    for (int s = 0; s < MS; ++s) atomicMax(&cmsh[s], fenc(sc[s]));
  }
  __syncthreads();
  if (threadIdx.x < MS) atomicMax(&colmax[threadIdx.x], cmsh[threadIdx.x]);
}

// ---------------- image pass 2: LDS-accumulated routed scatter ----------
// w[i, idx] = exp(sc[i,idx] - colmaxraw[idx])  (softmax shift/sum cancel)
__global__ __launch_bounds__(256) void kimg2(const float* __restrict__ image,
                                             const int* __restrict__ idxArr,
                                             const float* __restrict__ rowsc,
                                             const unsigned* __restrict__ colmax,
                                             float* __restrict__ seg,
                                             float* __restrict__ counts) {
  extern __shared__ float smem[];            // MS*DV seg + MS counts
  float* segSh = smem;
  float* cntSh = smem + MS * DV;
  int lane = threadIdx.x & 31, wave = threadIdx.x >> 5;
  for (int i = threadIdx.x; i < MS * DV + MS; i += 256) smem[i] = 0.0f;
  __syncthreads();

  for (int it = 0; it < 64; ++it) {
    int r = blockIdx.x * 512 + it * 8 + wave;
    const float* row = image + (size_t)r * DV;
    float x[24]; float ss = 0.0f;
#pragma unroll
    for (int j = 0; j < 24; ++j) { x[j] = row[lane + 32 * j]; ss += x[j] * x[j]; }
    ss = wredsum(ss);
    float inv = 1.0f / fmaxf(sqrtf(ss), 1e-12f);
    int slot = idxArr[r];
    float wq = expf(rowsc[r] - fdec(colmax[slot]));
    float* dst = segSh + (size_t)slot * DV;
#pragma unroll
    for (int j = 0; j < 24; ++j)
      atomicAdd(&dst[lane + 32 * j], wq * inv * x[j]);
    if (lane == 0) atomicAdd(&cntSh[slot], 1.0f);
  }
  __syncthreads();
  for (int i = threadIdx.x; i < MS * DV; i += 256) {
    float v = segSh[i];
    if (v != 0.0f) atomicAdd(&seg[i], v);
  }
  if (threadIdx.x < MS) atomicAdd(&counts[threadIdx.x], cntSh[threadIdx.x]);
}

// ---------------- finalize cache: momentum blend + normalize ------------
__global__ __launch_bounds__(256) void kcache(const float* __restrict__ cache,
                                              const float* __restrict__ seg,
                                              const float* __restrict__ counts,
                                              float* __restrict__ outc) {
  __shared__ float red[8];
  __shared__ float tot;
  int j = blockIdx.x, t = threadIdx.x;
  float cnt = counts[j];
  float u[3]; float ss = 0.0f;
#pragma unroll
  for (int e = 0; e < 3; ++e) {
    int col = t + 256 * e;
    float c0 = cache[(size_t)j * DV + col];
    float v = (cnt > 0.0f) ? 0.8f * c0 + 0.2f * seg[(size_t)j * DV + col] : c0;
    u[e] = v; ss += v * v;
  }
  ss = wredsum(ss);
  if ((t & 31) == 0) red[t >> 5] = ss;
  __syncthreads();
  if (t == 0) {
    float s = 0.0f;
    for (int i = 0; i < 8; ++i) s += red[i];
    tot = s;
  }
  __syncthreads();
  float inv = 1.0f / fmaxf(sqrtf(tot), 1e-12f);
#pragma unroll
  for (int e = 0; e < 3; ++e)
    outc[(size_t)j * DV + t + 256 * e] = u[e] * inv;
}

// ------------------------------------------------------------------------
extern "C" void kernel_launch(void* const* d_in, const int* in_sizes, int n_in,
                              void* d_out, int out_size, void* d_ws, size_t ws_size,
                              hipStream_t stream) {
  const float* text  = (const float*)d_in[0];
  const float* image = (const float*)d_in[1];
  const float* wex   = (const float*)d_in[2];
  const float* cache = (const float*)d_in[3];
  float* out = (float*)d_out;

  char* ws = (char*)d_ws;
  auto al = [](size_t x) { return (x + 255) & ~(size_t)255; };
  size_t o = 0;
  unsigned short* catbf = (unsigned short*)(ws + o); o = al(o + (size_t)CTOK * K2 * 2);
  unsigned short* wbf   = (unsigned short*)(ws + o); o = al(o + (size_t)DV * K2 * 2);
  int*      idxArr = (int*)     (ws + o); o = al(o + (size_t)NTOK * 4);
  float*    rowsc  = (float*)   (ws + o); o = al(o + (size_t)NTOK * 4);
  unsigned* colmax = (unsigned*)(ws + o); o = al(o + MS * 4);
  float*    seg    = (float*)   (ws + o); o = al(o + (size_t)MS * DV * 4);
  float*    counts = (float*)   (ws + o); o = al(o + MS * 4);

  float* loss = out + (size_t)CTOK * DV;
  float* cacheOut = loss + 1;

  kinit <<<1, 256, 0, stream>>>(seg, counts, loss, colmax);
  kconvw<<<(DV * K2 + 255) / 256, 256, 0, stream>>>(wex, wbf);
  kread <<<CTOK / 8, 256, 0, stream>>>(text, cache, catbf);
  kgemm <<<CTOK / 128, 256, 0, stream>>>(catbf, wbf, text, out);
  kloss <<<CTOK / 8, 256, 0, stream>>>(out, text, loss);
  kimg1 <<<NTOK / 8, 256, 0, stream>>>(image, cache, idxArr, rowsc, colmax);
  kimg2 <<<NTOK / 512, 256, (MS * DV + MS) * sizeof(float), stream>>>(
      image, idxArr, rowsc, colmax, seg, counts);
  kcache<<<MS, 256, 0, stream>>>(cache, seg, counts, cacheOut);
}
